// ExtractionNet_4733053960309
// MI455X (gfx1250) — matrix operand order repaired and checked by reading, not compiled, not run on hardware
//
#include <hip/hip_runtime.h>
#include <hip/hip_bf16.h>

typedef __attribute__((ext_vector_type(16))) __bf16 bf16x16;
typedef __attribute__((ext_vector_type(8)))  __bf16 bf16x8;
typedef __attribute__((ext_vector_type(8)))  float  f32x8;

#define T_    4
#define E_S_  4
#define E_SH_ 4
#define D_    1024
#define H_    2048
#define B_    4096
#define NG    8     // per-task gate width (E_SH + E_S)
#define NSG   20    // shared gate width (E_SH + T*E_S)

#define BM 128
#define BN 128
#define BK 32
#define APITCH 40   // ushort pitch (80B) -> 16B aligned ds_read_b128, conflict-free
#define WPITCH 40

__device__ __forceinline__ unsigned short f2bf(float f) {
  unsigned u = __float_as_uint(f);
  u += 0x7FFFu + ((u >> 16) & 1u);   // round-to-nearest-even
  return (unsigned short)(u >> 16);
}

__global__ void cvt_f32_bf16(const float* __restrict__ in,
                             unsigned short* __restrict__ out, int n) {
  int i = blockIdx.x * 256 + threadIdx.x;
  if (i < n) out[i] = f2bf(in[i]);
}

__global__ void zero_f32(float* __restrict__ p, int n) {
  int i = blockIdx.x * 256 + threadIdx.x;
  if (i < n) p[i] = 0.0f;
}

// One-time weight prep: fp32 [K][N] (per matrix) -> bf16 transposed [N][K].
// 64x64 tiles via LDS; fully coalesced loads and stores.
__global__ __launch_bounds__(256) void wprep(const float* __restrict__ in,
                                             unsigned short* __restrict__ out,
                                             int K, int N) {
  const long mat = blockIdx.z;
  in  += mat * (long)K * N;
  out += mat * (long)K * N;
  __shared__ unsigned short t[64][65];
  const int k0 = blockIdx.y * 64, n0 = blockIdx.x * 64;
  const int r = threadIdx.x >> 2;          // 0..63
  const int c = (threadIdx.x & 3) * 16;    // 0,16,32,48
  {
    const float* src = in + (long)(k0 + r) * N + n0 + c;
#pragma unroll
    for (int i = 0; i < 16; ++i) t[r][c + i] = f2bf(src[i]);
  }
  __syncthreads();
  {
    union { unsigned short s[16]; uint4 v[2]; } u;
#pragma unroll
    for (int i = 0; i < 16; ++i) u.s[i] = t[c + i][r];
    uint4* dst = (uint4*)(out + (long)(n0 + r) * K + k0 + c);
    dst[0] = u.v[0];
    dst[1] = u.v[1];
  }
}

// One 64-thread block per (t, b): logits = x . W[:,e] + bias, then softmax.
__global__ void gate_kernel(const float* __restrict__ X, long xStrideT,
                            const float* __restrict__ W, long wStrideT,
                            const float* __restrict__ bias, long bStrideT,
                            float* __restrict__ out, int E) {
  const int b = blockIdx.x, t = blockIdx.y;
  const float* x  = X + (long)t * xStrideT + (long)b * D_;
  const float* w  = W + (long)t * wStrideT;
  const float* bi = bias + (long)t * bStrideT;
  float part[NSG];
  for (int e = 0; e < NSG; ++e) part[e] = 0.f;
  for (int d = threadIdx.x; d < D_; d += 64) {
    float xv = x[d];
    const float* wr = w + (long)d * E;
    for (int e = 0; e < E; ++e) part[e] += xv * wr[e];
  }
  __shared__ float red[64 * NSG];
  for (int e = 0; e < E; ++e) red[e * 64 + threadIdx.x] = part[e];
  __syncthreads();
  for (int s = 32; s > 0; s >>= 1) {
    if ((int)threadIdx.x < s)
      for (int e = 0; e < E; ++e)
        red[e * 64 + threadIdx.x] += red[e * 64 + threadIdx.x + s];
    __syncthreads();
  }
  if (threadIdx.x == 0) {
    float l[NSG], m = -1e30f;
    for (int e = 0; e < E; ++e) { l[e] = red[e * 64] + bi[e]; m = fmaxf(m, l[e]); }
    float s = 0.f;
    for (int e = 0; e < E; ++e) { l[e] = __expf(l[e] - m); s += l[e]; }
    float inv = 1.f / s;
    float* o = out + ((long)t * gridDim.x + b) * E;
    for (int e = 0; e < E; ++e) o[e] = l[e] * inv;
  }
}

// ---- GEMM core (BM=BN=128, BK=32, 256 thr = 8 waves, 2x4 WMMA tiles/wave,
// double-buffered LDS, two barriers per k-step, bf16 A and bf16 W^T [N][K])

struct Frags { bf16x16 aF[2]; bf16x16 bF[4]; };
struct StageRegs { uint4 a0, a1, w0, w1; };

__device__ __forceinline__ StageRegs fetch_tile(const unsigned short* __restrict__ A,
                                                const unsigned short* __restrict__ Wt,
                                                int m0, int n0, int k0, int K, int tid) {
  StageRegs r;
  const int row = tid >> 1;
  const int kf  = (tid & 1) * 16;
  const uint4* as = (const uint4*)(A + (long)(m0 + row) * K + k0 + kf);
  r.a0 = as[0]; r.a1 = as[1];
  const uint4* ws = (const uint4*)(Wt + (long)(n0 + row) * K + k0 + kf);
  r.w0 = ws[0]; r.w1 = ws[1];
  return r;
}

__device__ __forceinline__ void store_tile(const StageRegs& r,
                                           unsigned short* As, unsigned short* Ws, int tid) {
  const int row = tid >> 1;
  const int kf  = (tid & 1) * 16;
  uint4* ad = (uint4*)(As + row * APITCH + kf);
  ad[0] = r.a0; ad[1] = r.a1;
  uint4* wd = (uint4*)(Ws + row * WPITCH + kf);
  wd[0] = r.w0; wd[1] = r.w1;
}

__device__ __forceinline__ Frags load_frags(const unsigned short* As,
                                            const unsigned short* Ws,
                                            int wm, int wn, int lane) {
  Frags f;
  const int half = lane >> 4;     // A: K{0-7,16-23}/{8-15,24-31}; B: K{0-7,16-23}/{8-15,24-31}
  const int l15  = lane & 15;
#pragma unroll
  for (int mi = 0; mi < 2; ++mi) {
    const unsigned short* p = As + (wm * 32 + mi * 16 + l15) * APITCH + half * 8;
    bf16x8 x0 = *(const bf16x8*)(p);
    bf16x8 x1 = *(const bf16x8*)(p + 16);
    f.aF[mi] = __builtin_shufflevector(x0, x1, 0,1,2,3,4,5,6,7,8,9,10,11,12,13,14,15);
  }
#pragma unroll
  for (int ni = 0; ni < 4; ++ni) {
    const unsigned short* p = Ws + (wn * 64 + ni * 16 + l15) * WPITCH + half * 8;
    bf16x8 x0 = *(const bf16x8*)(p);
    bf16x8 x1 = *(const bf16x8*)(p + 16);
    f.bF[ni] = __builtin_shufflevector(x0, x1, 0,1,2,3,4,5,6,7,8,9,10,11,12,13,14,15);
  }
  return f;
}

#define GEMM_PROLOGUE()                                                        \
  __shared__ __align__(16) unsigned short As[2][BM * APITCH];                  \
  __shared__ __align__(16) unsigned short Ws[2][BN * WPITCH];                  \
  const int tid = threadIdx.x, lane = tid & 31, wid = tid >> 5;                \
  const int wm = wid & 3, wn = wid >> 2;                                       \
  const int m0 = blockIdx.y * BM, n0 = blockIdx.x * BN;                        \
  f32x8 acc[2][4];                                                             \
  _Pragma("unroll") for (int i = 0; i < 2; ++i)                                \
  _Pragma("unroll") for (int j = 0; j < 4; ++j)                                \
  _Pragma("unroll") for (int k = 0; k < 8; ++k) acc[i][j][k] = 0.f;            \
  StageRegs rg = fetch_tile(A, Wt, m0, n0, 0, K, tid);                         \
  int buf = 0;                                                                 \
  for (int k0 = 0; k0 < K; k0 += BK) {                                         \
    store_tile(rg, As[buf], Ws[buf], tid);                                     \
    __syncthreads();                                                           \
    if (k0 + BK < K) rg = fetch_tile(A, Wt, m0, n0, k0 + BK, K, tid);          \
    Frags f = load_frags(As[buf], Ws[buf], wm, wn, lane);                      \
    _Pragma("unroll") for (int mi = 0; mi < 2; ++mi)                           \
    _Pragma("unroll") for (int ni = 0; ni < 4; ++ni)                           \
      acc[mi][ni] = __builtin_amdgcn_wmma_f32_16x16x32_bf16(                   \
          false, f.aF[mi], false, f.bF[ni], (short)0, acc[mi][ni], false, false); \
    buf ^= 1;                                                                  \
    __syncthreads();                                                           \
  }                                                                            \
  const int half = lane >> 4, l15 = lane & 15;

// GEMM1: H = relu(A @ W^T' + b) stored bf16   (A:[M,K] bf16, Wt:[N,K] bf16)
__global__ __launch_bounds__(256) void gemm1_relu(
    const unsigned short* __restrict__ A, const unsigned short* __restrict__ Wt,
    const float* __restrict__ bias, unsigned short* __restrict__ Hout,
    int M, int N, int K) {
  GEMM_PROLOGUE()
#pragma unroll
  for (int mi = 0; mi < 2; ++mi)
#pragma unroll
    for (int ni = 0; ni < 4; ++ni) {
      const int col = n0 + wn * 64 + ni * 16 + l15;
      const float bv = bias[col];
      const int rb = m0 + wm * 32 + mi * 16 + half * 8;
#pragma unroll
      for (int j = 0; j < 8; ++j) {
        float h = fmaxf(acc[mi][ni][j] + bv, 0.f);
        Hout[(long)(rb + j) * N + col] = f2bf(h);
      }
    }
}

// GEMM2: val = H1 @ W2^T' + b2; accumulate gate-weighted into task & shared outputs.
__global__ __launch_bounds__(256) void gemm2_acc(
    const unsigned short* __restrict__ A, const unsigned short* __restrict__ Wt,
    const float* __restrict__ bias,
    const float* __restrict__ gw, const float* __restrict__ sgw,
    float* __restrict__ outTask, float* __restrict__ outShared,
    int M, int N, int K, int taskFirst, int taskCnt, int gidx, int sgidx) {
  GEMM_PROLOGUE()
#pragma unroll
  for (int mi = 0; mi < 2; ++mi)
#pragma unroll
    for (int ni = 0; ni < 4; ++ni) {
      const int col = n0 + wn * 64 + ni * 16 + l15;
      const float bv = bias[col];
      const int rb = m0 + wm * 32 + mi * 16 + half * 8;
#pragma unroll
      for (int j = 0; j < 8; ++j) {
        const int row = rb + j;
        const float val = acc[mi][ni][j] + bv;
        for (int tt = 0; tt < taskCnt; ++tt) {
          const int t = taskFirst + tt;
          const float g = gw[((long)t * B_ + row) * NG + gidx];
          float* o = outTask + ((long)t * B_ + row) * (long)N + col;
          *o += g * val;
        }
        const float gs = sgw[(long)row * NSG + sgidx];
        outShared[(long)row * N + col] += gs * val;
      }
    }
}

extern "C" void kernel_launch(void* const* d_in, const int* in_sizes, int n_in,
                              void* d_out, int out_size, void* d_ws, size_t ws_size,
                              hipStream_t stream) {
  const float* task_reps  = (const float*)d_in[0];   // [T,B,D]
  const float* shared_rep = (const float*)d_in[1];   // [B,D]
  const float* spec_w1    = (const float*)d_in[2];   // [T,E_S,D,H]
  const float* spec_b1    = (const float*)d_in[3];   // [T,E_S,H]
  const float* spec_w2    = (const float*)d_in[4];   // [T,E_S,H,D]
  const float* spec_b2    = (const float*)d_in[5];   // [T,E_S,D]
  const float* shr_w1     = (const float*)d_in[6];   // [E_SH,D,H]
  const float* shr_b1     = (const float*)d_in[7];   // [E_SH,H]
  const float* shr_w2     = (const float*)d_in[8];   // [E_SH,H,D]
  const float* shr_b2     = (const float*)d_in[9];   // [E_SH,D]
  const float* gate_w     = (const float*)d_in[10];  // [T,D,NG]
  const float* gate_b     = (const float*)d_in[11];  // [T,NG]
  const float* sgate_w    = (const float*)d_in[12];  // [D,NSG]
  const float* sgate_b    = (const float*)d_in[13];  // [NSG]

  // workspace layout (all ushort chunks are multiples of 16B)
  unsigned short* Abf_task = (unsigned short*)d_ws;                // T*B*D
  unsigned short* Abf_sh   = Abf_task + (size_t)T_ * B_ * D_;      // B*D
  unsigned short* H1       = Abf_sh + (size_t)B_ * D_;             // B*H (reused)
  unsigned short* Wt_s1    = H1 + (size_t)B_ * H_;                 // [T*E_S][H][D]
  unsigned short* Wt_s2    = Wt_s1 + (size_t)T_ * E_S_ * D_ * H_;  // [T*E_S][D][H]
  unsigned short* Wt_h1    = Wt_s2 + (size_t)T_ * E_S_ * D_ * H_;  // [E_SH][H][D]
  unsigned short* Wt_h2    = Wt_h1 + (size_t)E_SH_ * D_ * H_;      // [E_SH][D][H]
  float* gw  = (float*)(Wt_h2 + (size_t)E_SH_ * D_ * H_);          // [T,B,NG]
  float* sgw = gw + (size_t)T_ * B_ * NG;                          // [B,NSG]

  float* outTask   = (float*)d_out;                                // [T,B,D]
  float* outShared = outTask + (size_t)T_ * B_ * D_;               // [B,D]

  // 1) one-time weight convert + transpose to bf16 [N][K]
  wprep<<<dim3(H_ / 64, D_ / 64, T_ * E_S_), 256, 0, stream>>>(spec_w1, Wt_s1, D_, H_);
  wprep<<<dim3(D_ / 64, H_ / 64, T_ * E_S_), 256, 0, stream>>>(spec_w2, Wt_s2, H_, D_);
  wprep<<<dim3(H_ / 64, D_ / 64, E_SH_), 256, 0, stream>>>(shr_w1, Wt_h1, D_, H_);
  wprep<<<dim3(D_ / 64, H_ / 64, E_SH_), 256, 0, stream>>>(shr_w2, Wt_h2, H_, D_);

  // 2) convert activations to bf16
  {
    int n = T_ * B_ * D_;
    cvt_f32_bf16<<<dim3((n + 255) / 256), 256, 0, stream>>>(task_reps, Abf_task, n);
    n = B_ * D_;
    cvt_f32_bf16<<<dim3((n + 255) / 256), 256, 0, stream>>>(shared_rep, Abf_sh, n);
  }
  // 3) zero outputs (we accumulate into them)
  zero_f32<<<dim3((out_size + 255) / 256), 256, 0, stream>>>((float*)d_out, out_size);

  // 4) gates
  gate_kernel<<<dim3(B_, T_), 64, 0, stream>>>(task_reps, (long)B_ * D_,
                                               gate_w, (long)D_ * NG,
                                               gate_b, (long)NG, gw, NG);
  gate_kernel<<<dim3(B_, 1), 64, 0, stream>>>(shared_rep, 0, sgate_w, 0,
                                              sgate_b, 0, sgw, NSG);

  // 5) experts (sequential launches -> deterministic accumulation order)
  dim3 g1(H_ / BN, B_ / BM);   // 16 x 32
  dim3 g2(D_ / BN, B_ / BM);   //  8 x 32

  for (int e = 0; e < E_SH_; ++e) {  // shared experts feed all tasks + shared gate idx e
    gemm1_relu<<<g1, 256, 0, stream>>>(Abf_sh, Wt_h1 + (size_t)e * D_ * H_,
                                       shr_b1 + (size_t)e * H_, H1, B_, H_, D_);
    gemm2_acc<<<g2, 256, 0, stream>>>(H1, Wt_h2 + (size_t)e * H_ * D_,
                                      shr_b2 + (size_t)e * D_, gw, sgw,
                                      outTask, outShared, B_, D_, H_,
                                      /*taskFirst*/0, /*taskCnt*/T_,
                                      /*gidx*/e, /*sgidx*/e);
  }
  for (int t = 0; t < T_; ++t)
    for (int e = 0; e < E_S_; ++e) {
      size_t xe = (size_t)t * E_S_ + e;
      gemm1_relu<<<g1, 256, 0, stream>>>(Abf_task + (size_t)t * B_ * D_,
                                         Wt_s1 + xe * (size_t)D_ * H_,
                                         spec_b1 + xe * (size_t)H_, H1, B_, H_, D_);
      gemm2_acc<<<g2, 256, 0, stream>>>(H1, Wt_s2 + xe * (size_t)H_ * D_,
                                        spec_b2 + xe * (size_t)D_, gw, sgw,
                                        outTask, outShared, B_, D_, H_,
                                        /*taskFirst*/t, /*taskCnt*/1,
                                        /*gidx*/E_SH_ + e,
                                        /*sgidx*/E_SH_ + t * E_S_ + e);
    }
}
